// CombinedLoss_5789615915057
// MI455X (gfx1250) — compile-verified, matrix operand order checked by reading, not run on hardware
//
#include <hip/hip_runtime.h>
#include <cstdint>
#include <cstddef>

#define NUM_CLASSES 19
#define TILE 32
#define HALO 20           // SDF_STEPS
#define EXT  72           // TILE + 2*HALO
#define NPIX (EXT*EXT)    // 5184
#define CORE (TILE*TILE)  // 1024
#define IMG  512

typedef __attribute__((ext_vector_type(2)))  float    v2f;
typedef __attribute__((ext_vector_type(8)))  float    v8f;
typedef __attribute__((ext_vector_type(16))) _Float16 v16h;

// ---- CDNA5 async global->LDS copy (ASYNCcnt path) -------------------------
__device__ __forceinline__ void async_load_b32_to_lds(const void* gaddr, unsigned lds_off) {
  asm volatile("global_load_async_to_lds_b32 %0, %1, off"
               :: "v"(lds_off), "v"(gaddr) : "memory");
}
__device__ __forceinline__ void wait_asynccnt0() {
#if __has_builtin(__builtin_amdgcn_s_wait_asynccnt)
  __builtin_amdgcn_s_wait_asynccnt(0);
#else
  asm volatile("s_wait_asynccnt 0" ::: "memory");
#endif
}

// ---- WMMA 256->1 block reduction (executed by one full wave32) ------------
// D = A x Ones + C accumulates row-sums of A; every source value is counted
// in all 16 columns, so sum(all D slots) == 16 * sum(values). Layout-agnostic.
__device__ __forceinline__ float wmma_reduce256(const float* buf, int lane) {
  float s;
#if __has_builtin(__builtin_amdgcn_wmma_f32_16x16x4_f32)
  v8f c = {};
  v2f ones; ones.x = 1.0f; ones.y = 1.0f;
#pragma unroll
  for (int g = 0; g < 4; ++g) {           // 4 chunks x 64 f32 values
    v2f a; a.x = buf[g*64 + lane]; a.y = buf[g*64 + 32 + lane];
    c = __builtin_amdgcn_wmma_f32_16x16x4_f32(false, a, false, ones,
                                              (short)0, c, false, false);
  }
  s = c[0]+c[1]+c[2]+c[3]+c[4]+c[5]+c[6]+c[7];
#else
  // fallback: codegen-confirmed f16 WMMA (A holds the 256 values, rest zero)
  v16h a, ones;
#pragma unroll
  for (int j = 0; j < 16; ++j) ones[j] = (_Float16)1.0f;
#pragma unroll
  for (int j = 0; j < 16; ++j) a[j] = (_Float16)0.0f;
#pragma unroll
  for (int j = 0; j < 8; ++j)  a[j] = (_Float16)buf[j*32 + lane];
  v8f c = {};
  c = __builtin_amdgcn_wmma_f32_16x16x32_f16(false, a, false, ones,
                                             (short)0, c, false, false);
  s = c[0]+c[1]+c[2]+c[3]+c[4]+c[5]+c[6]+c[7];
#endif
#pragma unroll
  for (int off = 16; off > 0; off >>= 1) s += __shfl_xor(s, off, 32);
  return s * (1.0f/16.0f);
}

// accum layout in d_ws: [0..18] inter, [19..37] psum, [38..56] tcnt,
//                       [57] focal_sum, [58] boundary_sum
__global__ void zero_accum(float* a) { if (threadIdx.x < 64) a[threadIdx.x] = 0.0f; }

__global__ __launch_bounds__(256)
void fused_loss_kernel(const float* __restrict__ pred,
                       const int*   __restrict__ target,
                       float*       __restrict__ accum) {
  __shared__ unsigned s_mask[2][NPIX];                                  // 41472 B
  __shared__ __attribute__((aligned(16))) unsigned char s_dist[NUM_CLASSES*CORE]; // 19456 B
  __shared__ float s_cls[3*NUM_CLASSES];                                // 228 B

  const int tid = threadIdx.x;
  const int tx = blockIdx.x, ty = blockIdx.y, b = blockIdx.z;
  const int h0 = ty*TILE - HALO, w0 = tx*TILE - HALO;
  const int* tgtB = target + b*IMG*IMG;

  for (int i = tid; i < (NUM_CLASSES*CORE)/4; i += 256) ((unsigned*)s_dist)[i] = 0u;
  for (int i = tid; i < 3*NUM_CLASSES; i += 256) s_cls[i] = 0.0f;

  // ---------- Phase 1a: async halo-tile load of target into s_mask[1] ----
#pragma unroll
  for (int k = 0; k < 21; ++k) {          // ceil(5184/256)
    int idx = tid + k*256;
    int ii = idx < NPIX ? idx : NPIX-1;   // clamp so every lane issues a valid copy
    int y = ii / EXT, x = ii - y*EXT;
    int gh = h0 + y, gw = w0 + x;
    int ghc = gh < 0 ? 0 : (gh > IMG-1 ? IMG-1 : gh);
    int gwc = gw < 0 ? 0 : (gw > IMG-1 ? IMG-1 : gw);
    async_load_b32_to_lds(tgtB + ghc*IMG + gwc, (unsigned)(size_t)&s_mask[1][ii]);
  }
  wait_asynccnt0();
  __syncthreads();

  // ---------- Phase 1b: class id -> 19-bit one-hot bitmask ----------------
  for (int idx = tid; idx < NPIX; idx += 256) {
    int y = idx / EXT, x = idx - y*EXT;
    int gh = h0 + y, gw = w0 + x;
    bool valid = ((unsigned)gh < (unsigned)IMG) && ((unsigned)gw < (unsigned)IMG);
    unsigned cls = s_mask[1][idx] & 31u;
    s_mask[0][idx] = valid ? (1u << cls) : 0u;  // outside image == zero padding
  }
  __syncthreads();

  // ---------- Phase 1c: 20 dilation steps, all 19 classes as one u32 ------
  int cur = 0;
  for (int it = 0; it < 20; ++it) {
    const unsigned* src = s_mask[cur];
    unsigned* dst = s_mask[cur^1];
    const int base = it + 1;
    const int W = EXT - 2*base;           // shrinking valid region
    const int cells = W*W;
    for (int c0 = tid; c0 < cells; c0 += 256) {
      int yy = c0 / W;
      int y = base + yy, x = base + (c0 - yy*W);
      const unsigned* p = src + y*EXT + x;
      unsigned nm = p[-EXT-1] | p[-EXT] | p[-EXT+1]
                  | p[-1]               | p[1]
                  | p[EXT-1]  | p[EXT]  | p[EXT+1];   // 3x3, zero center
      dst[y*EXT + x] = nm;
    }
    __syncthreads();
    // signed event accumulation (matches dist += new_pixels*(i+1), u8 wraps)
#pragma unroll
    for (int j = 0; j < 4; ++j) {
      int pix = tid + j*256;
      int my = (pix >> 5) + HALO, mx = (pix & 31) + HALO;
      unsigned om = src[my*EXT+mx], nm = dst[my*EXT+mx];
      unsigned ch = om ^ nm;
      while (ch) {
        int c = __builtin_ctz(ch); ch &= ch - 1;
        unsigned char d = s_dist[c*CORE + pix];
        s_dist[c*CORE + pix] = ((nm >> c) & 1u) ? (unsigned char)(d + it + 1)
                                                : (unsigned char)(d - (it + 1));
      }
    }
    __syncthreads();
    cur ^= 1;
  }
  // never-reached pixels -> SDF_STEPS
#pragma unroll
  for (int j = 0; j < 4; ++j) {
    int pix = tid + j*256;
    int my = (pix >> 5) + HALO, mx = (pix & 31) + HALO;
    unsigned fm = s_mask[cur][my*EXT+mx];
#pragma unroll
    for (int c = 0; c < NUM_CLASSES; ++c)
      if (!((fm >> c) & 1u)) s_dist[c*CORE + pix] = (unsigned char)20;
  }
  __syncthreads();   // s_mask is dead; will be reused as reduction buffer

  // ---------- Phase 2: single streaming pass over pred --------------------
  const size_t plane = (size_t)IMG*IMG;
  const float* predB = pred + (size_t)b*NUM_CLASSES*plane;
  const int hC = ty*TILE, wC = tx*TILE;
  float psum[NUM_CLASSES];
#pragma unroll
  for (int c = 0; c < NUM_CLASSES; ++c) psum[c] = 0.0f;
  float fpart = 0.0f, bpart = 0.0f;

#pragma unroll
  for (int j = 0; j < 4; ++j) {
    int pix = tid + j*256;                     // lanes -> consecutive w (coalesced)
    int gh = hC + (pix >> 5), gw = wC + (pix & 31);
    const float* pp = predB + (size_t)gh*IMG + gw;
    float r[NUM_CLASSES];
#pragma unroll
    for (int c = 0; c < NUM_CLASSES; ++c) r[c] = pp[c*plane];
    float m = r[0];
#pragma unroll
    for (int c = 1; c < NUM_CLASSES; ++c) m = fmaxf(m, r[c]);
    float se = 0.0f;
#pragma unroll
    for (int c = 0; c < NUM_CLASSES; ++c) se += __expf(r[c] - m);
    float logs = __logf(se);
    int t = tgtB[gh*IMG + gw];
    float pt = 0.0f, lpt = 0.0f;
#pragma unroll
    for (int c = 0; c < NUM_CLASSES; ++c) {
      float lp = r[c] - m - logs;
      float p  = __expf(lp);
      psum[c] += p;
      bpart   += p * (float)s_dist[c*CORE + pix];
      bool is = (c == t);
      lpt = is ? lp : lpt;
      pt  = is ? p  : pt;
    }
    fpart += -(1.0f - pt)*(1.0f - pt)*lpt;
    atomicAdd(&s_cls[t], pt);                        // intersection (ds_add_f32)
    atomicAdd(&s_cls[2*NUM_CLASSES + t], 1.0f);      // target count
  }

  // per-class: wave32 butterfly then one LDS atomic per wave
#pragma unroll
  for (int c = 0; c < NUM_CLASSES; ++c) {
    float v = psum[c];
#pragma unroll
    for (int off = 16; off > 0; off >>= 1) v += __shfl_xor(v, off, 32);
    if ((tid & 31) == 0) atomicAdd(&s_cls[NUM_CLASSES + c], v);
  }

  // focal/boundary: WMMA block reduction (reuse dead mask LDS)
  float* red = (float*)&s_mask[0][0];
  red[tid]       = fpart;
  red[256 + tid] = bpart;
  __syncthreads();

  if (tid < 32) {   // full wave 0, EXEC all ones -> WMMA legal
    float fsum = wmma_reduce256(red, tid);
    float bsum = wmma_reduce256(red + 256, tid);
    if (tid == 0) {
      atomicAdd(&accum[3*NUM_CLASSES + 0], fsum);
      atomicAdd(&accum[3*NUM_CLASSES + 1], bsum);
    }
  }
  if (tid < NUM_CLASSES) {
    atomicAdd(&accum[tid],                 s_cls[tid]);
    atomicAdd(&accum[NUM_CLASSES + tid],   s_cls[NUM_CLASSES + tid]);
    atomicAdd(&accum[2*NUM_CLASSES + tid], s_cls[2*NUM_CLASSES + tid]);
  }
}

__global__ void finalize_kernel(const float* __restrict__ a, float* __restrict__ out) {
  if (threadIdx.x == 0) {
    float dsum = 0.0f;
    for (int c = 0; c < NUM_CLASSES; ++c)
      dsum += (2.0f*a[c] + 1.0f) / (a[NUM_CLASSES + c] + a[2*NUM_CLASSES + c] + 1.0f);
    float dice = 1.0f - dsum / (float)NUM_CLASSES;
    const float npix = 8.0f * 512.0f * 512.0f;
    float focal    = a[3*NUM_CLASSES + 0] / npix;
    float boundary = a[3*NUM_CLASSES + 1] / (npix * (float)NUM_CLASSES);
    out[0] = 0.3f*dice + 0.3f*focal + 0.1f*boundary;
    out[1] = dice;
    out[2] = focal;
    out[3] = boundary;
  }
}

extern "C" void kernel_launch(void* const* d_in, const int* in_sizes, int n_in,
                              void* d_out, int out_size, void* d_ws, size_t ws_size,
                              hipStream_t stream) {
  const float* pred   = (const float*)d_in[0];
  const int*   target = (const int*)d_in[1];
  float* accum = (float*)d_ws;
  float* out   = (float*)d_out;

  zero_accum<<<1, 64, 0, stream>>>(accum);
  dim3 grid(IMG/TILE, IMG/TILE, 8);   // 16 x 16 x 8 = 2048 blocks, 256 thr (8 waves)
  fused_loss_kernel<<<grid, 256, 0, stream>>>(pred, target, accum);
  finalize_kernel<<<1, 32, 0, stream>>>(accum, out);
}
